// CVLearner_5669356835096
// MI455X (gfx1250) — compile-verified, matrix operand order checked by reading, not run on hardware
//
#include <hip/hip_runtime.h>

// ---------------------------------------------------------------------------
// Types for CDNA5 WMMA (wave32): V_WMMA_F32_16X16X32_BF16
// ---------------------------------------------------------------------------
typedef __attribute__((ext_vector_type(16))) __bf16 v16bf;
typedef __attribute__((ext_vector_type(8)))  float  v8f;

__device__ __forceinline__ unsigned short f32_to_bf16_bits(float f) {
  union { float f; unsigned u; } in; in.f = f;
  unsigned u = in.u;
  unsigned r = (u + 0x7FFFu + ((u >> 16) & 1u)) >> 16;  // round-to-nearest-even
  return (unsigned short)r;
}
__device__ __forceinline__ float bf16_bits_to_f32(unsigned short s) {
  union { unsigned u; float f; } c; c.u = ((unsigned)s) << 16; return c.f;
}

// A/B fragment = two contiguous 8-element (16-byte) K-runs per lane.
union FragU { uint4 q[2]; v16bf v; };

// ---------------------------------------------------------------------------
// conv4d as implicit GEMM:  Y[Cout,P] = Wbf[Cout,Kpad] x im2col(X)[K,P]
// One wave computes a 16(cout) x 16(pos) f32 tile; K in chunks of 64
// (2 x V_WMMA_F32_16X16X32_BF16 per staging round).
//  - KK (kernel width) is a template constant: tap decode is constant-divisor
//    magic; no exec branches anywhere in the K loop.
//  - Position decode: ONE div/mod chain for pos0, the other 15 positions are
//    derived with carry increments (kills the scalar-division prologue).
//  - Position base coords packed 4x8bit (+4 bias) in registers; per-element
//    tap offset is one packed v_add; linear index separable:
//    idx = tapIdx(lane,chunk) + posIdx[e](block).
//  - A (weights bf16, rows zero-padded to Kpad): 2x global b128 per sub-chunk.
//  - B (im2col): branchless gather -> LDS transposed [n][K] -> 2x ds b128.
// ---------------------------------------------------------------------------
template <bool SRCF32, int KK>
__global__ void __launch_bounds__(32)
conv4d_wmma(const void* __restrict__ Xv, const unsigned short* __restrict__ Wbf,
            const float* __restrict__ bias, float* __restrict__ Y,
            int Cin, int Cout,
            int I1, int I2, int I3, int I4,
            int O1, int O2, int O3, int O4,
            int s1, int s2, int s3, int s4, int pad,
            int P, int Ktot, int Kpad)
{
  constexpr int K2 = KK * KK;
  constexpr int K3 = K2 * KK;
  constexpr int K4 = K3 * KK;

  const int lane  = threadIdx.x;       // 0..31
  const int pos0  = blockIdx.x * 16;   // N tile
  const int cout0 = blockIdx.y * 16;   // M tile
  const int b     = blockIdx.z;
  const int bCin  = b * Cin;

  const float*          Xf = (const float*)Xv;
  const unsigned short* Xb = (const unsigned short*)Xv;

  // B tiles, transposed [n][K]; row stride 40 ushorts (80 B, 16B-aligned).
  __shared__ __align__(16) unsigned short lB0[16 * 40];
  __shared__ __align__(16) unsigned short lB1[16 * 40];

  // Per-position packed base coords (+4 bias per 8-bit field) and base linear
  // index. Uniform values -> SGPRs. Only pos0 is decoded with divisions; the
  // rest increment with carries.
  unsigned pcPack[16];
  int      pcIdx[16];
  {
    int o4 = pos0 % O4; int r = pos0 / O4;
    int o3 = r % O3;    r /= O3;
    int o2 = r % O2;    int o1 = r / O2;
#pragma unroll
    for (int e = 0; e < 16; ++e) {
      const int b1 = o1 * s1 - pad, b2 = o2 * s2 - pad;
      const int b3 = o3 * s3 - pad, b4 = o4 * s4 - pad;
      const unsigned pk = (unsigned)(b1 + 4) | ((unsigned)(b2 + 4) << 8) |
                          ((unsigned)(b3 + 4) << 16) | ((unsigned)(b4 + 4) << 24);
      const bool valid = (pos0 + e) < P;
      pcPack[e] = valid ? pk : (200u | (200u << 8) | (200u << 16) | (200u << 24));
      pcIdx[e]  = ((b1 * I2 + b2) * I3 + b3) * I4 + b4;  // masked when invalid
      // carry-increment to next position
      ++o4;
      if (o4 == O4) {
        o4 = 0; ++o3;
        if (o3 == O3) {
          o3 = 0; ++o2;
          if (o2 == O2) { o2 = 0; ++o1; }
        }
      }
    }
  }

  v8f acc = {};

  const int half = lane >> 4;   // 0 | 1
  const int mrow = lane & 15;   // M for A, N for B

  // Direct-from-global A fragment base (row = cout0+mrow, zero-padded to Kpad).
  const unsigned short* arow = Wbf + (size_t)(cout0 + mrow) * Kpad;

  for (int k0 = 0; k0 < Kpad; k0 += 64) {
    // ---- branchless im2col gather: lane owns K rows (k0+lane, k0+32+lane) ----
#pragma unroll
    for (int c = 0; c < 2; ++c) {
      unsigned short* lB = c ? lB1 : lB0;
      const int kk = k0 + c * 32 + lane;
      const bool okk = (kk < Ktot);
      // constant-divisor decode (values past Ktot produce garbage -> masked)
      unsigned ukk = (unsigned)kk;
      unsigned ci = ukk / K4; unsigned r = ukk - ci * K4;
      unsigned t1 = r / K3;   r -= t1 * K3;
      unsigned t2 = r / K2;   r -= t2 * K2;
      unsigned t3 = r / KK;   unsigned t4 = r - t3 * KK;
      const int tapIdx = ((((bCin + (int)ci) * I1 + (int)t1) * I2 + (int)t2) * I3 +
                          (int)t3) * I4 + (int)t4;
      const unsigned tp = t1 | (t2 << 8) | (t3 << 16) | (t4 << 24);
#pragma unroll
      for (int e = 0; e < 16; ++e) {
        unsigned sf = pcPack[e] + tp;  // packed per-dim coords (+4 bias)
        bool ok = okk &
                  ((unsigned)((sf & 255u) - 4u)         < (unsigned)I1) &
                  ((unsigned)(((sf >> 8) & 255u) - 4u)  < (unsigned)I2) &
                  ((unsigned)(((sf >> 16) & 255u) - 4u) < (unsigned)I3) &
                  ((unsigned)((sf >> 24) - 4u)          < (unsigned)I4);
        int idx  = tapIdx + pcIdx[e];
        int lidx = ok ? idx : 0;      // clamp -> unconditional load
        unsigned short bits;
        if (SRCF32) {
          float xv = Xf[lidx];
          bits = f32_to_bf16_bits(ok ? xv : 0.f);
        } else {
          unsigned short bv = Xb[lidx];
          bits = ok ? bv : (unsigned short)0;
        }
        lB[e * 40 + lane] = bits;
      }
    }
    __syncthreads();

    // prefetch next weight chunk while we do math (global_prefetch_b8)
    if (k0 + 64 < Kpad) __builtin_prefetch(arow + k0 + 64, 0, 1);

    // ---- 2 sub-chunks of K=32: A direct from global, B from LDS ----
#pragma unroll
    for (int c = 0; c < 2; ++c) {
      const unsigned short* abase = arow + k0 + c * 32;
      const unsigned short* lB = c ? lB1 : lB0;
      FragU a, bm;
      // lane's K runs: [8*half, 8*half+8) and [16+8*half, ...+8)
      a.q[0]  = *(const uint4*)(abase + 8 * half);
      a.q[1]  = *(const uint4*)(abase + 16 + 8 * half);
      bm.q[0] = *(const uint4*)(lB + mrow * 40 + 8 * half);
      bm.q[1] = *(const uint4*)(lB + mrow * 40 + 16 + 8 * half);
      acc = __builtin_amdgcn_wmma_f32_16x16x32_bf16(
          /*neg_a=*/false, a.v, /*neg_b=*/false, bm.v,
          /*c_mod=*/(short)0, acc, /*reuse_a=*/false, /*reuse_b=*/false);
    }
    __syncthreads();
  }

  // ---- store 16x16 f32 tile (+bias). C/D: VGPR r, lane L -> M=r+8*(L>>4), N=L&15
  const int pos = pos0 + mrow;
  if (pos < P) {
#pragma unroll
    for (int r = 0; r < 8; ++r) {
      int co = cout0 + r + half * 8;  // Cout is always a multiple of 16
      Y[((size_t)b * Cout + co) * P + pos] = acc[r] + bias[co];
    }
  }
}

// ---------------------------------------------------------------------------
// Per-layer weight conversion f32 -> bf16, rows zero-padded Ktot -> Kpad.
// ---------------------------------------------------------------------------
__global__ void __launch_bounds__(256)
weights_to_bf16(const float* __restrict__ src, unsigned short* __restrict__ dst,
                int Ktot, int Kpad, int total)
{
  int idx = blockIdx.x * 256 + threadIdx.x;
  if (idx >= total) return;
  int co = idx / Kpad;
  int kk = idx - co * Kpad;
  dst[idx] = (kk < Ktot) ? f32_to_bf16_bits(src[(size_t)co * Ktot + kk])
                         : (unsigned short)0;
}

// ---------------------------------------------------------------------------
// GroupNorm (GROUPS=4, population var, eps=1e-5) + ReLU:
// reads f32 conv output, writes bf16 activations for the next conv.
// One block per (group, batch); group span is contiguous (Cg*P floats).
// ---------------------------------------------------------------------------
__global__ void __launch_bounds__(256)
groupnorm_relu_bf16(const float* __restrict__ src, unsigned short* __restrict__ dst,
                    const float* __restrict__ gamma, const float* __restrict__ beta,
                    int C, int P, int Cg)
{
  const int g = blockIdx.x;
  const int b = blockIdx.y;
  const int tid = threadIdx.x;
  const size_t off = ((size_t)b * C + (size_t)g * Cg) * P;
  const float* base = src + off;
  const int n = Cg * P;

  __shared__ float sSum[256];
  __shared__ float sSq[256];
  float s = 0.f, q = 0.f;
  for (int i = tid; i < n; i += 256) {
    float v = base[i];
    s += v; q += v * v;
  }
  sSum[tid] = s; sSq[tid] = q;
  __syncthreads();
#pragma unroll
  for (int o = 128; o > 0; o >>= 1) {
    if (tid < o) { sSum[tid] += sSum[tid + o]; sSq[tid] += sSq[tid + o]; }
    __syncthreads();
  }
  __shared__ float sMean, sRstd;
  if (tid == 0) {
    float m = sSum[0] / (float)n;
    float v = sSq[0] / (float)n - m * m;
    sMean = m;
    sRstd = rsqrtf(v + 1e-5f);
  }
  __syncthreads();
  const float m = sMean, r = sRstd;
  for (int c = 0; c < Cg; ++c) {                 // channel-outer: no divisions
    const float ga = gamma[g * Cg + c] * r;
    const float be = beta[g * Cg + c];
    const float* sc = base + (size_t)c * P;
    unsigned short* dc = dst + off + (size_t)c * P;
    for (int i = tid; i < P; i += 256) {
      float v = (sc[i] - m) * ga + be;
      v = v > 0.f ? v : 0.f;
      dc[i] = f32_to_bf16_bits(v);
    }
  }
}

// ---------------------------------------------------------------------------
// Mean pooling over spatial dim (bf16 in, f32 out).
// ---------------------------------------------------------------------------
__global__ void __launch_bounds__(256)
pool_mean(const unsigned short* __restrict__ x, float* __restrict__ pooled,
          int P, int total)
{
  int idx = blockIdx.x * 256 + threadIdx.x;
  if (idx >= total) return;
  const unsigned short* p = x + (size_t)idx * P;
  float s = 0.f;
  for (int i = 0; i < P; ++i) s += bf16_bits_to_f32(p[i]);
  pooled[idx] = s / (float)P;
}

// ---------------------------------------------------------------------------
// MLP head: h = relu(pooled @ mw1^T + mb1); out = h @ mw2^T + mb2 (f32, tiny).
// ---------------------------------------------------------------------------
__global__ void __launch_bounds__(256)
mlp_head(const float* __restrict__ pooled,
         const float* __restrict__ mw1, const float* __restrict__ mb1,
         const float* __restrict__ mw2, const float* __restrict__ mb2,
         float* __restrict__ out)
{
  __shared__ float h[8 * 128];
  const int tid = threadIdx.x;
  for (int idx = tid; idx < 8 * 128; idx += 256) {
    int b = idx >> 7, j = idx & 127;
    const float* pr = pooled + b * 128;
    const float* wr = mw1 + j * 128;
    float s = mb1[j];
#pragma unroll 4
    for (int t = 0; t < 128; ++t) s += pr[t] * wr[t];
    h[idx] = s > 0.f ? s : 0.f;
  }
  __syncthreads();
  if (tid < 16) {
    int b = tid >> 1, o = tid & 1;
    const float* hr = h + b * 128;
    const float* wr = mw2 + o * 128;
    float s = mb2[o];
#pragma unroll 4
    for (int t = 0; t < 128; ++t) s += hr[t] * wr[t];
    out[b * 2 + o] = s;
  }
}

// ---------------------------------------------------------------------------
// Host-side pipeline. Input dict order:
//   d_in[0] = corr (8,8,24,24,24,24) f32
//   d_in[1+4i]=w_i, d_in[2+4i]=b_i, d_in[3+4i]=g_i, d_in[4+4i]=be_i (i=0..8)
//   d_in[37]=mw1, d_in[38]=mb1, d_in[39]=mw2, d_in[40]=mb2
// ---------------------------------------------------------------------------
extern "C" void kernel_launch(void* const* d_in, const int* in_sizes, int n_in,
                              void* d_out, int out_size, void* d_ws, size_t ws_size,
                              hipStream_t stream) {
  (void)in_sizes; (void)n_in; (void)out_size; (void)ws_size;

  static const int LAY[9][5] = {
      {8, 16, 5, 2, 2}, {16, 16, 3, 1, 1}, {16, 32, 3, 2, 2},
      {32, 32, 3, 1, 1}, {32, 32, 3, 1, 1}, {32, 64, 3, 2, 2},
      {64, 64, 3, 1, 1}, {64, 64, 3, 1, 1}, {64, 128, 3, 1, 1}};
  const int B = 8;

  // Workspace layout (bytes):
  //   [0,   12M)  convBuf: f32 conv output, reused every layer (needs 10.6 MiB)
  //   [12M, 20M)  actA:    bf16 activations ping (needs 5.1 MiB)
  //   [20M, 28M)  actB:    bf16 activations pong
  //   [28M, 30M)  wbuf:    bf16 padded weights (needs <= 1.3 MiB)
  //   [30M, ...)  pooled:  f32 [8,128]
  char* ws = (char*)d_ws;
  float*          convBuf = (float*)ws;
  unsigned short* actA    = (unsigned short*)(ws + (12u << 20));
  unsigned short* actB    = (unsigned short*)(ws + (20u << 20));
  unsigned short* wbuf    = (unsigned short*)(ws + (28u << 20));
  float*          pooled  = (float*)(ws + (30u << 20));

  int I1 = 24, I2 = 24, I3 = 24, I4 = 24;
  int lastP = 0, lastC = 0;
  const unsigned short* prevAct = nullptr;

  for (int i = 0; i < 9; ++i) {
    const int Cin = LAY[i][0], Cout = LAY[i][1], k = LAY[i][2];
    const int qs = LAY[i][3], ks = LAY[i][4];
    const int pad = k / 2;
    const int s1 = qs, s2 = qs, s3 = ks, s4 = ks;
    const int O1 = (I1 + 2 * pad - k) / s1 + 1;
    const int O2 = (I2 + 2 * pad - k) / s2 + 1;
    const int O3 = (I3 + 2 * pad - k) / s3 + 1;
    const int O4 = (I4 + 2 * pad - k) / s4 + 1;
    const int P = O1 * O2 * O3 * O4;
    const int Ktot = Cin * k * k * k * k;
    const int Kpad = (Ktot + 63) & ~63;

    // weights -> bf16, rows zero-padded to Kpad (kills all K guards in conv)
    {
      int total = Cout * Kpad;
      weights_to_bf16<<<(total + 255) / 256, 256, 0, stream>>>(
          (const float*)d_in[1 + 4 * i], wbuf, Ktot, Kpad, total);
    }

    dim3 grid((P + 15) / 16, Cout / 16, B);
    if (i == 0) {
      conv4d_wmma<true, 5><<<grid, 32, 0, stream>>>(
          d_in[0], wbuf, (const float*)d_in[2 + 4 * i], convBuf,
          Cin, Cout, I1, I2, I3, I4, O1, O2, O3, O4,
          s1, s2, s3, s4, pad, P, Ktot, Kpad);
    } else {
      conv4d_wmma<false, 3><<<grid, 32, 0, stream>>>(
          prevAct, wbuf, (const float*)d_in[2 + 4 * i], convBuf,
          Cin, Cout, I1, I2, I3, I4, O1, O2, O3, O4,
          s1, s2, s3, s4, pad, P, Ktot, Kpad);
    }

    unsigned short* actCur = (i & 1) ? actB : actA;
    groupnorm_relu_bf16<<<dim3(4, B), 256, 0, stream>>>(
        convBuf, actCur, (const float*)d_in[3 + 4 * i],
        (const float*)d_in[4 + 4 * i], Cout, P, Cout / 4);

    prevAct = actCur;
    I1 = O1; I2 = O2; I3 = O3; I4 = O4;
    lastP = P; lastC = Cout;
  }

  pool_mean<<<(B * lastC + 255) / 256, 256, 0, stream>>>(
      prevAct, pooled, lastP, B * lastC);

  mlp_head<<<1, 256, 0, stream>>>(
      pooled, (const float*)d_in[37], (const float*)d_in[38],
      (const float*)d_in[39], (const float*)d_in[40], (float*)d_out);
}
